// DARNN_44152263803491
// MI455X (gfx1250) — matrix operand order repaired and checked by reading, not compiled, not run on hardware
//
#include <hip/hip_runtime.h>

typedef _Float16 h16;
typedef __attribute__((ext_vector_type(16))) _Float16 v16h;
typedef __attribute__((ext_vector_type(8)))  _Float16 v8h;
typedef __attribute__((ext_vector_type(8)))  float    v8f;

constexpr int Bsz = 2048;
constexpr int Wn  = 64;
constexpr int Fn  = 128;
constexpr int Hn  = 128;

// ---------------- WMMA fragment helpers (16x16x32 f16 -> f32) ----------------
static __device__ __forceinline__ v16h loadA_frag(const h16* A, int ldk, int kbase) {
  int lane = threadIdx.x & 31;
  int m = lane & 15;
  int koff = (lane >> 4) << 3;              // 0 or 8
  const h16* base = A + m * ldk + kbase + koff;
  v8h a = *(const v8h*)(base);
  v8h b = *(const v8h*)(base + 16);
  return __builtin_shufflevector(a, b, 0,1,2,3,4,5,6,7,8,9,10,11,12,13,14,15);
}

static __device__ __forceinline__ v16h loadB_frag(const h16* W, int ldk, int ntile, int ktile) {
  int lane = threadIdx.x & 31;
  int n = ntile * 16 + (lane & 15);
  int kb = ktile * 32 + ((lane >> 4) << 3);
  v8h lo = *(const v8h*)(W + n * ldk + kb);
  v8h hi = *(const v8h*)(W + n * ldk + kb + 16);
  return __builtin_shufflevector(lo, hi, 0,1,2,3,4,5,6,7,8,9,10,11,12,13,14,15);
}

static __device__ __forceinline__ v8f wmma32(v16h a, v16h b, v8f c) {
  return __builtin_amdgcn_wmma_f32_16x16x32_f16(false, a, false, b, (short)0, c, false, false);
}

static __device__ __forceinline__ float sigf(float x) { return 1.f / (1.f + __expf(-x)); }

// ---------------- weight conversion kernels (f32 [N][Kfull] -> f16 [N][Ksub]) ----------------
__global__ void k_conv(const float* __restrict__ src, h16* __restrict__ dst,
                       int N, int Kfull, int c0, int Ksub) {
  int i = blockIdx.x * 256 + threadIdx.x;
  if (i < N * Ksub) {
    int n = i / Ksub, k = i - n * Ksub;
    dst[i] = (h16)src[n * Kfull + c0 + k];
  }
}

__global__ void k_conv2(const float* __restrict__ A, const float* __restrict__ B,
                        h16* __restrict__ dst, int N, int Ka, int Kb) {
  int i = blockIdx.x * 256 + threadIdx.x;
  int K = Ka + Kb;
  if (i < N * K) {
    int n = i / K, k = i - n * K;
    dst[i] = (h16)(k < Ka ? A[n * Ka + k] : B[n * Kb + (k - Ka)]);
  }
}

// ---------------- P[b][f][j] = sum_t X[b,t,f] * ia_W1[j,t]  (per-batch 128x64x64 GEMM) ---------
__global__ __launch_bounds__(128) void k_precompP(const float* __restrict__ X,
                                                  const h16* __restrict__ Wx,  // [64][64] f16
                                                  h16* __restrict__ P) {       // [B][F][64]
  __shared__ h16 At[Fn * Wn];  // A = X[b].T : [f][t]
  int b = blockIdx.x;
  int tid = threadIdx.x;
  const float* Xb = X + (size_t)b * Wn * Fn;
  #pragma unroll 1
  for (int t = 0; t < Wn; ++t)
    At[tid * Wn + t] = (h16)Xb[t * Fn + tid];   // f = tid
  __syncthreads();
  int wv = tid >> 5, lane = tid & 31;
  int mhi = (lane >> 4) << 3, nl = lane & 15;
  // wave owns M-tiles {2wv, 2wv+1}; per M-tile: all 4 N-tiles as independent chains.
  #pragma unroll 1
  for (int mtl = 0; mtl < 2; ++mtl) {
    int mt = wv * 2 + mtl;
    v8f acc[4] = {{}, {}, {}, {}};
    #pragma unroll
    for (int kt = 0; kt < 2; ++kt) {
      v16h a = loadA_frag(At + mt * 16 * Wn, Wn, kt * 32);
      #pragma unroll
      for (int nt = 0; nt < 4; ++nt)
        acc[nt] = wmma32(a, loadB_frag(Wx, Wn, nt, kt), acc[nt]);
    }
    #pragma unroll
    for (int nt = 0; nt < 4; ++nt) {
      h16* Pp = P + (size_t)b * Fn * 64 + (size_t)(mt * 16) * 64 + nt * 16 + nl;
      #pragma unroll
      for (int r = 0; r < 8; ++r)
        Pp[(r + mhi) * 64] = (h16)acc[nt][r];
    }
  }
}

// ---------------- encoder scan: 1 block = 16 batch rows, 4 waves, 64 steps ----------------
__global__ __launch_bounds__(128) void k_encoder(
    const float* __restrict__ X,
    const h16* __restrict__ Whc,     // [64][256]  (ia_W1 h|c part)
    const h16* __restrict__ Wg,      // [512][256] (enc_Wih|enc_Whh)
    const h16* __restrict__ P,       // [B][F][64]
    const float* __restrict__ ia_b1, const float* __restrict__ ia_W2, const float* __restrict__ ia_b2,
    const float* __restrict__ enc_bih, const float* __restrict__ enc_bhh,
    h16* __restrict__ Hs)            // [W][B][H]
{
  __shared__ h16 hc16[16 * 256];   // [h | c] f16 (A for qh GEMM)
  __shared__ h16 xh16[16 * 256];   // [x~ | h] f16 (A for LSTM GEMM)
  __shared__ float c32[16 * 128];  // cell state f32
  __shared__ float g32[16 * 512];  // gates; overlays: e32 = g32[0:2048], qh32 = g32[2048:3072]
  __shared__ float rmax[16], rinv[16];
  __shared__ float w2s[64];        // ia_W2 cached
  float* e32  = g32;               // [16][128] attention scores (dead before g32 written)
  float* qh32 = g32 + 16 * 128;    // [16][64]

  int tid = threadIdx.x;
  int wv = tid >> 5, lane = tid & 31;
  int mhi = (lane >> 4) << 3, nl = lane & 15;
  int bg0 = blockIdx.x * 16;

  for (int i = tid; i < 16 * 256; i += 128) hc16[i] = (h16)0.f;
  for (int i = tid; i < 16 * 128; i += 128) c32[i] = 0.f;
  if (tid < 64) w2s[tid] = ia_W2[tid];
  __syncthreads();

  #pragma unroll 1
  for (int t = 0; t < Wn; ++t) {
    // --- qh = [h|c] @ Whc.T + ia_b1 : [16,256]x[256,64]; 4 K-parity chains ---
    {
      v16h af[8];
      #pragma unroll
      for (int kt = 0; kt < 8; ++kt) af[kt] = loadA_frag(hc16, 256, kt * 32);
      v8f acc[4] = {{}, {}, {}, {}};
      #pragma unroll
      for (int kt = 0; kt < 2; ++kt) {
        #pragma unroll
        for (int j = 0; j < 4; ++j)
          acc[j] = wmma32(af[4 * kt + j], loadB_frag(Whc, 256, wv, 4 * kt + j), acc[j]);
      }
      int n = wv * 16 + nl;
      float bias = ia_b1[n];
      #pragma unroll
      for (int r = 0; r < 8; ++r)
        qh32[(r + mhi) * 64 + n] = acc[0][r] + acc[1][r] + acc[2][r] + acc[3][r] + bias;
    }
    __syncthreads();
    // --- e[b][f] = ia_b2 + sum_j ia_W2[j]*tanh(P[b,f,j]+qh[b,j]); f = tid ---
    {
      float b2 = ia_b2[0];
      #pragma unroll 1
      for (int b = 0; b < 16; ++b) {
        const h16* Pp = P + ((size_t)(bg0 + b) * Fn + tid) * 64;
        if (b < 15) __builtin_prefetch(Pp + (size_t)Fn * 64, 0, 3);  // next row, L2-warm
        const float* q = qh32 + b * 64;
        float s = b2;
        #pragma unroll
        for (int c = 0; c < 4; ++c) {
          v16h pv = *(const v16h*)(Pp + c * 16);
          #pragma unroll
          for (int e = 0; e < 16; ++e)
            s += w2s[c * 16 + e] * tanhf((float)pv[e] + q[c * 16 + e]);
        }
        e32[b * 128 + tid] = s;
      }
    }
    __syncthreads();
    if (tid < 16) {  // softmax over f (per batch row)
      float m = -1e30f;
      for (int f = 0; f < 128; ++f) m = fmaxf(m, e32[tid * 128 + f]);
      float s = 0.f;
      for (int f = 0; f < 128; ++f) s += __expf(e32[tid * 128 + f] - m);
      rmax[tid] = m; rinv[tid] = 1.f / s;
    }
    __syncthreads();
    // --- x_tilde = alpha * x_t ; pack A = [x~ | h] ---
    #pragma unroll 1
    for (int b = 0; b < 16; ++b) {
      float alpha = __expf(e32[b * 128 + tid] - rmax[b]) * rinv[b];
      float xv = X[((size_t)(bg0 + b) * Wn + t) * Fn + tid];
      xh16[b * 256 + tid]       = (h16)(alpha * xv);
      xh16[b * 256 + 128 + tid] = hc16[b * 256 + tid];
    }
    __syncthreads();
    // --- LSTM gates g = [x~|h] @ Wg.T + (bih+bhh) : [16,256]x[256,512]; 4 N-tiles/pass ---
    {
      v16h af[8];
      #pragma unroll
      for (int kt = 0; kt < 8; ++kt) af[kt] = loadA_frag(xh16, 256, kt * 32);
      #pragma unroll 1
      for (int q2 = 0; q2 < 2; ++q2) {
        int ntb = wv * 8 + q2 * 4;
        v8f acc[4] = {{}, {}, {}, {}};
        #pragma unroll
        for (int kt = 0; kt < 8; ++kt) {
          #pragma unroll
          for (int j = 0; j < 4; ++j)
            acc[j] = wmma32(af[kt], loadB_frag(Wg, 256, ntb + j, kt), acc[j]);
        }
        #pragma unroll
        for (int j = 0; j < 4; ++j) {
          int n = (ntb + j) * 16 + nl;
          float bias = enc_bih[n] + enc_bhh[n];
          #pragma unroll
          for (int r = 0; r < 8; ++r)
            g32[(r + mhi) * 512 + n] = acc[j][r] + bias;
        }
      }
    }
    __syncthreads();
    // --- elementwise gates: thread owns (b = tid>>3, 16 h's) ---
    {
      int b = tid >> 3, h0 = (tid & 7) * 16;
      h16* Hrow = Hs + ((size_t)t * Bsz + bg0 + b) * Hn;
      #pragma unroll 4
      for (int e = 0; e < 16; ++e) {
        int hh = h0 + e;
        float gi = g32[b * 512 + hh],       gf = g32[b * 512 + 128 + hh];
        float gg = g32[b * 512 + 256 + hh], go = g32[b * 512 + 384 + hh];
        float c = sigf(gf) * c32[b * 128 + hh] + sigf(gi) * tanhf(gg);
        c32[b * 128 + hh] = c;
        float hnew = sigf(go) * tanhf(c);
        hc16[b * 256 + hh]       = (h16)hnew;
        hc16[b * 256 + 128 + hh] = (h16)c;
        Hrow[hh] = (h16)hnew;
      }
    }
    __syncthreads();
  }
}

// ---------------- PH[w][b][:] = Hs[w][b][:] @ ta_W1[:, :H].T ----------------
__global__ __launch_bounds__(128) void k_precompPH(const h16* __restrict__ Hs,
                                                   const h16* __restrict__ Wph,  // [128][128]
                                                   h16* __restrict__ PH) {
  int w = blockIdx.y;
  int bg0 = blockIdx.x * 16;
  int tid = threadIdx.x, wv = tid >> 5, lane = tid & 31;
  int mhi = (lane >> 4) << 3, nl = lane & 15;
  const h16* A = Hs + ((size_t)w * Bsz + bg0) * Hn;   // [16][128] global
  v16h a[4];
  #pragma unroll
  for (int kt = 0; kt < 4; ++kt) a[kt] = loadA_frag(A, 128, kt * 32);
  int nt0 = wv * 2, nt1 = nt0 + 1;
  // 4 chains: 2 N-tiles x 2 K-parity
  v8f c00 = {}, c01 = {}, c10 = {}, c11 = {};
  #pragma unroll
  for (int kp = 0; kp < 2; ++kp) {
    c00 = wmma32(a[2 * kp],     loadB_frag(Wph, 128, nt0, 2 * kp),     c00);
    c10 = wmma32(a[2 * kp],     loadB_frag(Wph, 128, nt1, 2 * kp),     c10);
    c01 = wmma32(a[2 * kp + 1], loadB_frag(Wph, 128, nt0, 2 * kp + 1), c01);
    c11 = wmma32(a[2 * kp + 1], loadB_frag(Wph, 128, nt1, 2 * kp + 1), c11);
  }
  h16* O0 = PH + ((size_t)w * Bsz + bg0) * Hn + nt0 * 16 + nl;
  h16* O1 = PH + ((size_t)w * Bsz + bg0) * Hn + nt1 * 16 + nl;
  #pragma unroll
  for (int r = 0; r < 8; ++r) {
    O0[(size_t)(r + mhi) * Hn] = (h16)(c00[r] + c01[r]);
    O1[(size_t)(r + mhi) * Hn] = (h16)(c10[r] + c11[r]);
  }
}

// ---------------- decoder scan ----------------
__global__ __launch_bounds__(128) void k_decoder(
    const h16* __restrict__ Hs, const h16* __restrict__ PH,
    const h16* __restrict__ Wqd,   // [128][256] (ta_W1 d|ds part)
    const h16* __restrict__ Wdec,  // [512][128] (dec_Whh)
    const h16* __restrict__ Wl2,   // [128][256] (l2_W)
    const float* __restrict__ ta_b1, const float* __restrict__ ta_W2, const float* __restrict__ ta_b2,
    const float* __restrict__ dec_Wih, const float* __restrict__ dec_bih, const float* __restrict__ dec_bhh,
    const float* __restrict__ l1_W, const float* __restrict__ l1_b,
    const float* __restrict__ l2_b, const float* __restrict__ l3_W, const float* __restrict__ l3_b,
    float* __restrict__ out)
{
  __shared__ h16 dds16[16 * 256];  // [d | ds] f16
  __shared__ h16 A2[16 * 256];     // [ct | d_new] f16
  __shared__ float ds32[16 * 128]; // cell state f32
  __shared__ float g32[16 * 512];  // gates; overlays below (barrier-ordered)
  __shared__ float ytv[16], outv[16];
  __shared__ float w2t[128];       // ta_W2 cached
  float* qd32 = g32;               // [16][128] (dead before g32 written)
  float* sc   = g32 + 16 * 128;    // [16][64]  scores -> betas
  float* ct32 = g32 + 16 * 128 + 16 * 64;  // [16][128]
  float* o32  = g32;               // [16][128] (written after gates consumed g32)

  int tid = threadIdx.x, wv = tid >> 5, lane = tid & 31;
  int mhi = (lane >> 4) << 3, nl = lane & 15;
  int bg0 = blockIdx.x * 16;

  for (int i = tid; i < 16 * 256; i += 128) dds16[i] = (h16)0.f;
  for (int i = tid; i < 16 * 128; i += 128) ds32[i] = 0.f;
  if (tid < 16) outv[tid] = 0.f;
  w2t[tid] = ta_W2[tid];
  __syncthreads();

  #pragma unroll 1
  for (int t = 0; t < Wn; ++t) {
    // A fragments of [d|ds] (qd GEMM; first 4 = d-part reused by dec LSTM)
    v16h af[8];
    #pragma unroll
    for (int kt = 0; kt < 8; ++kt) af[kt] = loadA_frag(dds16, 256, kt * 32);
    // --- qd = [d|ds] @ Wqd.T + ta_b1 : 2 N-tiles x 2 K-parity = 4 chains ---
    {
      int nt0 = wv * 2, nt1 = nt0 + 1;
      v8f c00 = {}, c01 = {}, c10 = {}, c11 = {};
      #pragma unroll
      for (int kp = 0; kp < 4; ++kp) {
        c00 = wmma32(af[2 * kp],     loadB_frag(Wqd, 256, nt0, 2 * kp),     c00);
        c10 = wmma32(af[2 * kp],     loadB_frag(Wqd, 256, nt1, 2 * kp),     c10);
        c01 = wmma32(af[2 * kp + 1], loadB_frag(Wqd, 256, nt0, 2 * kp + 1), c01);
        c11 = wmma32(af[2 * kp + 1], loadB_frag(Wqd, 256, nt1, 2 * kp + 1), c11);
      }
      int n0 = nt0 * 16 + nl, n1 = nt1 * 16 + nl;
      float bias0 = ta_b1[n0], bias1 = ta_b1[n1];
      #pragma unroll
      for (int r = 0; r < 8; ++r) {
        qd32[(r + mhi) * 128 + n0] = c00[r] + c01[r] + bias0;
        qd32[(r + mhi) * 128 + n1] = c10[r] + c11[r] + bias1;
      }
    }
    __syncthreads();
    // --- scores l[w][b]: thread handles 8 (w,b) pairs ---
    {
      float b2 = ta_b2[0];
      #pragma unroll 1
      for (int pp = 0; pp < 8; ++pp) {
        int p = tid * 8 + pp;
        int wi = p >> 4, b = p & 15;
        const h16* ph = PH + ((size_t)wi * Bsz + bg0 + b) * Hn;
        const float* q = qd32 + b * 128;
        float s = b2;
        #pragma unroll 2
        for (int c = 0; c < 8; ++c) {
          v16h pv = *(const v16h*)(ph + c * 16);
          #pragma unroll
          for (int e = 0; e < 16; ++e)
            s += w2t[c * 16 + e] * tanhf((float)pv[e] + q[c * 16 + e]);
        }
        sc[b * 64 + wi] = s;
      }
    }
    __syncthreads();
    if (tid < 16) {  // softmax over w; overwrite sc with beta
      float m = -1e30f;
      for (int w = 0; w < 64; ++w) m = fmaxf(m, sc[tid * 64 + w]);
      float s = 0.f;
      for (int w = 0; w < 64; ++w) s += __expf(sc[tid * 64 + w] - m);
      float inv = 1.f / s;
      for (int w = 0; w < 64; ++w) sc[tid * 64 + w] = __expf(sc[tid * 64 + w] - m) * inv;
    }
    __syncthreads();
    // --- ct = sum_w beta * Hs[w] : thread owns (b, 16 h's), vector loads of Hs ---
    {
      int b = tid >> 3, h0 = (tid & 7) * 16;
      float acc[16];
      #pragma unroll
      for (int e = 0; e < 16; ++e) acc[e] = 0.f;
      #pragma unroll 1
      for (int wi = 0; wi < 64; ++wi) {
        float beta = sc[b * 64 + wi];
        v16h hrow = *(const v16h*)(Hs + ((size_t)wi * Bsz + bg0 + b) * Hn + h0);
        #pragma unroll
        for (int e = 0; e < 16; ++e) acc[e] += beta * (float)hrow[e];
      }
      #pragma unroll
      for (int e = 0; e < 16; ++e) {
        ct32[b * 128 + h0 + e] = acc[e];
        A2[b * 256 + h0 + e] = (h16)acc[e];
      }
    }
    __syncthreads();
    if (tid < 16) {  // yt = [out, ct] @ l1_W.T + l1_b
      float s = l1_b[0] + l1_W[0] * outv[tid];
      for (int h = 0; h < 128; ++h) s += l1_W[1 + h] * ct32[tid * 128 + h];
      ytv[tid] = s;
    }
    __syncthreads();
    // --- decoder LSTM: g = yt*dec_Wih + d @ dec_Whh.T + biases; 4 N-tiles/pass ---
    #pragma unroll 1
    for (int q2 = 0; q2 < 2; ++q2) {
      int ntb = wv * 8 + q2 * 4;
      v8f acc[4] = {{}, {}, {}, {}};
      #pragma unroll
      for (int kt = 0; kt < 4; ++kt) {
        #pragma unroll
        for (int j = 0; j < 4; ++j)
          acc[j] = wmma32(af[kt], loadB_frag(Wdec, 128, ntb + j, kt), acc[j]);
      }
      #pragma unroll
      for (int j = 0; j < 4; ++j) {
        int n = (ntb + j) * 16 + nl;
        float bias = dec_bih[n] + dec_bhh[n];
        float wih = dec_Wih[n];
        #pragma unroll
        for (int r = 0; r < 8; ++r)
          g32[(r + mhi) * 512 + n] = acc[j][r] + bias + ytv[r + mhi] * wih;
      }
    }
    __syncthreads();
    // --- gates ---
    {
      int b = tid >> 3, h0 = (tid & 7) * 16;
      #pragma unroll 4
      for (int e = 0; e < 16; ++e) {
        int hh = h0 + e;
        float gi = g32[b * 512 + hh],       gf = g32[b * 512 + 128 + hh];
        float gg = g32[b * 512 + 256 + hh], go = g32[b * 512 + 384 + hh];
        float c = sigf(gf) * ds32[b * 128 + hh] + sigf(gi) * tanhf(gg);
        ds32[b * 128 + hh] = c;
        float d = sigf(go) * tanhf(c);
        dds16[b * 256 + hh]       = (h16)d;
        dds16[b * 256 + 128 + hh] = (h16)c;
        A2[b * 256 + 128 + hh]    = (h16)d;
      }
    }
    __syncthreads();
    // --- o = [ct|d_new] @ l2_W.T + l2_b : 2 N-tiles x 2 K-parity = 4 chains ---
    {
      v16h a2f[8];
      #pragma unroll
      for (int kt = 0; kt < 8; ++kt) a2f[kt] = loadA_frag(A2, 256, kt * 32);
      int nt0 = wv * 2, nt1 = nt0 + 1;
      v8f c00 = {}, c01 = {}, c10 = {}, c11 = {};
      #pragma unroll
      for (int kp = 0; kp < 4; ++kp) {
        c00 = wmma32(a2f[2 * kp],     loadB_frag(Wl2, 256, nt0, 2 * kp),     c00);
        c10 = wmma32(a2f[2 * kp],     loadB_frag(Wl2, 256, nt1, 2 * kp),     c10);
        c01 = wmma32(a2f[2 * kp + 1], loadB_frag(Wl2, 256, nt0, 2 * kp + 1), c01);
        c11 = wmma32(a2f[2 * kp + 1], loadB_frag(Wl2, 256, nt1, 2 * kp + 1), c11);
      }
      int n0 = nt0 * 16 + nl, n1 = nt1 * 16 + nl;
      float bias0 = l2_b[n0], bias1 = l2_b[n1];
      #pragma unroll
      for (int r = 0; r < 8; ++r) {
        o32[(r + mhi) * 128 + n0] = c00[r] + c01[r] + bias0;
        o32[(r + mhi) * 128 + n1] = c10[r] + c11[r] + bias1;
      }
    }
    __syncthreads();
    if (tid < 16) {  // out = sigmoid(o @ l3_W.T + l3_b)
      float s = l3_b[0];
      for (int h = 0; h < 128; ++h) s += o32[tid * 128 + h] * l3_W[h];
      outv[tid] = sigf(s);
    }
    __syncthreads();
  }
  if (tid < 16) out[bg0 + tid] = outv[tid];
}

// ---------------- host launcher ----------------
extern "C" void kernel_launch(void* const* d_in, const int* in_sizes, int n_in,
                              void* d_out, int out_size, void* d_ws, size_t ws_size,
                              hipStream_t stream) {
  (void)in_sizes; (void)n_in; (void)out_size; (void)ws_size;
  const float* X       = (const float*)d_in[0];
  const float* ia_W1   = (const float*)d_in[1];
  const float* ia_b1   = (const float*)d_in[2];
  const float* ia_W2   = (const float*)d_in[3];
  const float* ia_b2   = (const float*)d_in[4];
  const float* enc_Wih = (const float*)d_in[5];
  const float* enc_Whh = (const float*)d_in[6];
  const float* enc_bih = (const float*)d_in[7];
  const float* enc_bhh = (const float*)d_in[8];
  const float* ta_W1   = (const float*)d_in[9];
  const float* ta_b1   = (const float*)d_in[10];
  const float* ta_W2   = (const float*)d_in[11];
  const float* ta_b2   = (const float*)d_in[12];
  const float* dec_Wih = (const float*)d_in[13];
  const float* dec_Whh = (const float*)d_in[14];
  const float* dec_bih = (const float*)d_in[15];
  const float* dec_bhh = (const float*)d_in[16];
  const float* l1_W    = (const float*)d_in[17];
  const float* l1_b    = (const float*)d_in[18];
  const float* l2_W    = (const float*)d_in[19];
  const float* l2_b    = (const float*)d_in[20];
  const float* l3_W    = (const float*)d_in[21];
  const float* l3_b    = (const float*)d_in[22];

  // f16 workspace layout (~101 MB total; P/Hs/PH are L2-resident)
  h16* ws   = (h16*)d_ws;
  h16* wX   = ws;                         // [64][64]
  h16* wHC  = wX  + 64 * 64;              // [64][256]
  h16* wEnc = wHC + 64 * 256;             // [512][256]
  h16* wPHm = wEnc + 512 * 256;           // [128][128]
  h16* wQD  = wPHm + 128 * 128;           // [128][256]
  h16* wDec = wQD + 128 * 256;            // [512][128]
  h16* wL2  = wDec + 512 * 128;           // [128][256]
  h16* P    = wL2 + 128 * 256;            // [B][F][64]
  h16* Hsb  = P   + (size_t)Bsz * Fn * 64;  // [W][B][H]
  h16* PHb  = Hsb + (size_t)Wn * Bsz * Hn;  // [W][B][H]

  // convert weights to f16 ([N][K] row-major; B-fragments read contiguously)
  k_conv <<<(64 * 64   + 255) / 256, 256, 0, stream>>>(ia_W1,  wX,  64, 320, 0, 64);
  k_conv <<<(64 * 256  + 255) / 256, 256, 0, stream>>>(ia_W1,  wHC, 64, 320, 64, 256);
  k_conv2<<<(512 * 256 + 255) / 256, 256, 0, stream>>>(enc_Wih, enc_Whh, wEnc, 512, 128, 128);
  k_conv <<<(128 * 128 + 255) / 256, 256, 0, stream>>>(ta_W1,  wPHm, 128, 384, 0, 128);
  k_conv <<<(128 * 256 + 255) / 256, 256, 0, stream>>>(ta_W1,  wQD, 128, 384, 128, 256);
  k_conv <<<(512 * 128 + 255) / 256, 256, 0, stream>>>(dec_Whh, wDec, 512, 128, 0, 128);
  k_conv <<<(128 * 256 + 255) / 256, 256, 0, stream>>>(l2_W,   wL2, 128, 256, 0, 256);

  k_precompP <<<Bsz, 128, 0, stream>>>(X, wX, P);
  k_encoder  <<<Bsz / 16, 128, 0, stream>>>(X, wHC, wEnc, P, ia_b1, ia_W2, ia_b2,
                                            enc_bih, enc_bhh, Hsb);
  k_precompPH<<<dim3(Bsz / 16, Wn), 128, 0, stream>>>(Hsb, wPHm, PHb);
  k_decoder  <<<Bsz / 16, 128, 0, stream>>>(Hsb, PHb, wQD, wDec, wL2,
                                            ta_b1, ta_W2, ta_b2,
                                            dec_Wih, dec_bih, dec_bhh,
                                            l1_W, l1_b, l2_b, l3_W, l3_b,
                                            (float*)d_out);
}
